// NemotronHMoE_91164975825163
// MI455X (gfx1250) — matrix operand order repaired and checked by reading, not compiled, not run on hardware
//
#include <hip/hip_runtime.h>
#include <hip/hip_bf16.h>
#include <math.h>

// ---------------------------------------------------------------------------
// Problem constants (from reference)
// ---------------------------------------------------------------------------
#define TT   4096      // tokens = B*S
#define HH   2048      // hidden
#define EE   32        // experts
#define II   1024      // expert intermediate
#define SSI  4096      // shared intermediate
#define KSEL 4         // experts per token
#define NGRP 4         // groups
#define TKG  2         // topk groups
#define SCALE_F 2.5f

typedef __attribute__((ext_vector_type(16))) __bf16 v16bf;
typedef __attribute__((ext_vector_type(8)))  __bf16 v8bf;
typedef __attribute__((ext_vector_type(8)))  float  v8f;

// 16-byte payload type matching the async-copy builtin's parameter type
typedef int vec4i __attribute__((vector_size(16)));
#define GAS __attribute__((address_space(1)))   // global
#define LAS __attribute__((address_space(3)))   // LDS

// CDNA5 async global->LDS copy (ASYNCcnt-tracked), guarded so the file still
// compiles if this toolchain doesn't declare the builtin.
#if __has_builtin(__builtin_amdgcn_global_load_async_to_lds_b128)
#define ASYNC_LDS 1
#else
#define ASYNC_LDS 0
#endif

__device__ __forceinline__ void async_wait0() {
#if ASYNC_LDS
#if __has_builtin(__builtin_amdgcn_s_wait_asynccnt)
  __builtin_amdgcn_s_wait_asynccnt(0);
#else
  asm volatile("s_wait_asynccnt 0" ::: "memory");
#endif
#endif
}

// ---------------------------------------------------------------------------
// fp32 -> bf16 elementwise converter (8 elements / thread, b128 stores)
// ---------------------------------------------------------------------------
__global__ __launch_bounds__(256) void f2bf_kernel(const float* __restrict__ in,
                                                   __bf16* __restrict__ out, int n8) {
  int i = blockIdx.x * 256 + threadIdx.x;
  if (i >= n8) return;
  const float4* p = reinterpret_cast<const float4*>(in) + (size_t)i * 2;
  float4 a = p[0], b = p[1];
  v8bf o;
  o[0] = (__bf16)a.x; o[1] = (__bf16)a.y; o[2] = (__bf16)a.z; o[3] = (__bf16)a.w;
  o[4] = (__bf16)b.x; o[5] = (__bf16)b.y; o[6] = (__bf16)b.z; o[7] = (__bf16)b.w;
  *reinterpret_cast<v8bf*>(out + (size_t)i * 8) = o;
}

__global__ void zero_counts_kernel(int* counts) { counts[threadIdx.x] = 0; }

// ---------------------------------------------------------------------------
// Router: one wave32 per token. lane e computes logits[t,e]; lane 0 does the
// 32-element group-limited top-k (exact reference semantics, lowest-index ties)
// ---------------------------------------------------------------------------
__global__ __launch_bounds__(32) void router_kernel(
    const float* __restrict__ x, const float* __restrict__ gw,
    const float* __restrict__ eb, int* __restrict__ counts,
    int* __restrict__ tokens, float* __restrict__ wts) {
  const int t = blockIdx.x;
  const int lane = threadIdx.x;           // lane == expert id
  const float* __restrict__ h = x + (size_t)t * HH;

  float acc = 0.f;
  #pragma unroll 8
  for (int k = 0; k < HH; ++k) acc = fmaf(h[k], gw[(size_t)k * EE + lane], acc);
  float score = 1.f / (1.f + __expf(-acc));

  __shared__ float s_scores[EE];
  __shared__ float s_sfc[EE];
  s_scores[lane] = score;
  s_sfc[lane]    = score + eb[lane];
  __syncthreads();

  if (lane == 0) {
    // group scores = sum of top-2 within each group of 8
    float gs[NGRP];
    for (int g = 0; g < NGRP; ++g) {
      float m1 = -1e30f, m2 = -1e30f;
      for (int j = 0; j < EE / NGRP; ++j) {
        float v = s_sfc[g * (EE / NGRP) + j];
        if (v > m1) { m2 = m1; m1 = v; } else if (v > m2) { m2 = v; }
      }
      gs[g] = m1 + m2;
    }
    // top TKG groups
    bool gsel[NGRP] = {false, false, false, false};
    for (int it = 0; it < TKG; ++it) {
      int best = 0; float bv = -1e30f;
      for (int g = 0; g < NGRP; ++g)
        if (!gsel[g] && gs[g] > bv) { bv = gs[g]; best = g; }
      gsel[best] = true;
    }
    // masked candidate values (masked-out -> 0.0 exactly as reference)
    float v[EE];
    for (int e = 0; e < EE; ++e) v[e] = gsel[e >> 3] ? s_sfc[e] : 0.0f;
    // top K experts, collect unbiased scores
    int idx[KSEL]; float tw[KSEL];
    for (int it = 0; it < KSEL; ++it) {
      int best = 0; float bv = -1e30f;
      for (int e = 0; e < EE; ++e)
        if (v[e] > bv) { bv = v[e]; best = e; }
      idx[it] = best; tw[it] = s_scores[best]; v[best] = -1e30f;
    }
    float s = tw[0] + tw[1] + tw[2] + tw[3] + 1e-20f;
    float sc = SCALE_F / s;
    for (int k = 0; k < KSEL; ++k) {
      int e = idx[k];
      int pos = atomicAdd(&counts[e], 1);
      tokens[(size_t)e * TT + pos] = t;
      wts[(size_t)e * TT + pos]    = tw[k] * sc;
    }
  }
}

__global__ void prefix_kernel(const int* __restrict__ counts, int* __restrict__ offs) {
  if (threadIdx.x == 0) {
    int s = 0;
    for (int e = 0; e < EE; ++e) { offs[e] = s; s += counts[e]; }
    offs[EE] = s;
  }
}

// ---------------------------------------------------------------------------
// bf16 WMMA GEMM: C[M,N] = A[M,K] * B[K,N]  (A,B row-major bf16, f32 accum)
// Workgroup tile 64x128 (8 waves, each wave 2x2 of 16x16 wmma tiles), BK=32.
// Double-buffered LDS, ONE barrier per K-step; tiles staged either by CDNA5
// async global->LDS copies (ASYNCcnt) or by register round-trip fallback.
//   EPI 0: shared-up    -> bf16 relu^2 store
//   EPI 1: shared-down  -> f32 plain store (initializes d_out)
//   EPI 2: routed-up    -> gathered A rows (token list), bf16 relu^2 store
//   EPI 3: routed-down  -> weighted atomicAdd f32 into d_out
// ---------------------------------------------------------------------------
__device__ __forceinline__ v16bf pack16(v8bf lo, v8bf hi) {
  v16bf r;
  #pragma unroll
  for (int i = 0; i < 8; ++i) { r[i] = lo[i]; r[8 + i] = hi[i]; }
  return r;
}

template <int EPI>
__global__ __launch_bounds__(256) void gemm_bf16_kernel(
    const __bf16* __restrict__ A, int lda,
    const __bf16* __restrict__ B, int ldb,
    int M, int N, int Kd,
    float* __restrict__ outF, __bf16* __restrict__ outB, int ldo,
    const int* __restrict__ counts, const int* __restrict__ offs,
    const int* __restrict__ tokens, const float* __restrict__ wts) {
  constexpr int BM = 64, BN = 128, BK = 32;
  const int tid  = threadIdx.x;
  const int lane = tid & 31;
  const int wid  = tid >> 5;
  const int wm   = wid & 1;    // 2 M-subtiles of 32
  const int wn   = wid >> 1;   // 4 N-subtiles of 32

  const int e = blockIdx.z;
  int Me = M;
  const __bf16* Ab = A;
  const __bf16* Bb = B;
  const int* rmap = nullptr;
  int orow = 0;
  if (EPI >= 2) {
    Me = counts[e];
    if ((int)blockIdx.x * BM >= Me) return;
    Bb = B + (size_t)e * Kd * N;
    if (EPI == 2) { rmap = tokens + (size_t)e * TT; orow = offs[e]; }
    else          { Ab = A + (size_t)offs[e] * lda; }
  }

  __shared__ __align__(16) __bf16 As[2][BM * BK];   // 2 x 64x32
  __shared__ __align__(16) __bf16 Bs[2][BK * BN];   // 2 x 32x128

  // ---- hoisted global row pointers -------------------------------------
  const int ra = tid >> 2;             // A tile row   (64 rows, 4 thr/row)
  const int ca = (tid & 3) * 8;        // A k-chunk
  int grow = blockIdx.x * BM + ra;
  const __bf16* aRow;
  if (EPI == 2) {
    int rr = grow < Me ? grow : (Me - 1);
    aRow = Ab + (size_t)rmap[rr] * lda;
  } else if (EPI == 3) {
    int rr = grow < Me ? grow : (Me - 1);
    aRow = Ab + (size_t)rr * lda;
  } else {
    aRow = Ab + (size_t)grow * lda;
  }
  aRow += ca;

  const int rb = tid >> 3;             // B tile row   (32 rows, 8 thr/row)
  const int cb = (tid & 7) * 16;       // B n-chunk (16 elems = 2 x b128)
  const __bf16* bRow = Bb + (size_t)rb * ldb + (size_t)blockIdx.y * BN + cb;

  // stage one K-tile into LDS buffer `buf`
  auto load_tile = [&](int k0, int buf) {
    const __bf16* ap = aRow + k0;
    const __bf16* bp = bRow + (size_t)k0 * ldb;
#if ASYNC_LDS
    __builtin_amdgcn_global_load_async_to_lds_b128(
        (GAS vec4i*)ap, (LAS vec4i*)&As[buf][ra * BK + ca], 0, 0);
    __builtin_amdgcn_global_load_async_to_lds_b128(
        (GAS vec4i*)bp, (LAS vec4i*)&Bs[buf][rb * BN + cb], 0, 0);
    __builtin_amdgcn_global_load_async_to_lds_b128(
        (GAS vec4i*)(bp + 8), (LAS vec4i*)&Bs[buf][rb * BN + cb + 8], 0, 0);
#else
    *reinterpret_cast<v8bf*>(&As[buf][ra * BK + ca]) =
        *reinterpret_cast<const v8bf*>(ap);
    *reinterpret_cast<v8bf*>(&Bs[buf][rb * BN + cb]) =
        *reinterpret_cast<const v8bf*>(bp);
    *reinterpret_cast<v8bf*>(&Bs[buf][rb * BN + cb + 8]) =
        *reinterpret_cast<const v8bf*>(bp + 8);
    __builtin_prefetch(ap + BK, 0, 1);
    __builtin_prefetch(bp + (size_t)BK * ldb, 0, 1);
#endif
  };

  v8f acc[2][2];
  #pragma unroll
  for (int i = 0; i < 2; ++i)
    #pragma unroll
    for (int j = 0; j < 2; ++j)
      #pragma unroll
      for (int r = 0; r < 8; ++r) acc[i][j][r] = 0.f;

  const int half = lane >> 4;
  const int ml   = lane & 15;
  const int nk   = Kd / BK;

  // prologue: tile 0 into buffer 0
  load_tile(0, 0);
  async_wait0();
  __syncthreads();

  for (int i = 0; i < nk; ++i) {
    const int cur = i & 1;
    if (i + 1 < nk) load_tile((i + 1) * BK, cur ^ 1);

    // A fragments: 16-bit A 16x32 layout (lane half selects K phase)
    v16bf af[2];
    #pragma unroll
    for (int im = 0; im < 2; ++im) {
      int row = wm * 32 + im * 16 + ml;
      v8bf lo = *reinterpret_cast<const v8bf*>(&As[cur][row * BK + half * 8]);
      v8bf hi = *reinterpret_cast<const v8bf*>(&As[cur][row * BK + 16 + half * 8]);
      af[im] = pack16(lo, hi);
    }
    // B fragments: N = lane % 16, K in the same order as A (half*8+s, 16+half*8+s)
    v16bf bfr[2];
    #pragma unroll
    for (int in_ = 0; in_ < 2; ++in_) {
      int n0 = wn * 32 + in_ * 16;
      #pragma unroll
      for (int s = 0; s < 8; ++s) {
        bfr[in_][s]     = Bs[cur][(half * 8 + s) * BN + n0 + ml];
        bfr[in_][8 + s] = Bs[cur][(16 + half * 8 + s) * BN + n0 + ml];
      }
    }
    #pragma unroll
    for (int im = 0; im < 2; ++im)
      #pragma unroll
      for (int in_ = 0; in_ < 2; ++in_)
        acc[im][in_] = __builtin_amdgcn_wmma_f32_16x16x32_bf16(
            false, af[im], false, bfr[in_], (short)0, acc[im][in_], false, false);

    // next tile must be resident AND everyone done reading `cur` before the
    // next iteration overwrites it: single barrier per K-step.
    async_wait0();
    __syncthreads();
  }

  // ---- epilogue: C/D layout: VGPR r, lanes 0-15 -> M=r, lanes 16-31 -> M=8+r
  #pragma unroll
  for (int im = 0; im < 2; ++im) {
    #pragma unroll
    for (int in_ = 0; in_ < 2; ++in_) {
      int rbase = blockIdx.x * BM + wm * 32 + im * 16 + half * 8;
      int col   = blockIdx.y * BN + wn * 32 + in_ * 16 + ml;
      v8f a = acc[im][in_];
      #pragma unroll
      for (int r = 0; r < 8; ++r) {
        int row = rbase + r;
        float val = a[r];
        if (EPI == 0) {
          float rv = val > 0.f ? val * val : 0.f;
          outB[(size_t)row * ldo + col] = (__bf16)rv;
        } else if (EPI == 1) {
          outF[(size_t)row * ldo + col] = val;
        } else if (EPI == 2) {
          if (row < Me) {
            float rv = val > 0.f ? val * val : 0.f;
            outB[(size_t)(orow + row) * ldo + col] = (__bf16)rv;
          }
        } else {  // EPI == 3
          if (row < Me) {
            float w = wts[(size_t)e * TT + row];
            int   t = tokens[(size_t)e * TT + row];
            atomicAdd(&outF[(size_t)t * ldo + col], val * w);
          }
        }
      }
    }
  }
}

// ---------------------------------------------------------------------------
// Host-side orchestration
// ---------------------------------------------------------------------------
static inline size_t align256(size_t x) { return (x + 255) & ~(size_t)255; }

extern "C" void kernel_launch(void* const* d_in, const int* in_sizes, int n_in,
                              void* d_out, int out_size, void* d_ws, size_t ws_size,
                              hipStream_t stream) {
  (void)in_sizes; (void)n_in; (void)out_size; (void)ws_size;
  const float* x    = (const float*)d_in[0];  // [T,H]
  const float* gw   = (const float*)d_in[1];  // [H,E]
  const float* eb   = (const float*)d_in[2];  // [E]
  const float* Wu   = (const float*)d_in[3];  // [E,H,I]
  const float* Wd   = (const float*)d_in[4];  // [E,I,H]
  const float* Wsu  = (const float*)d_in[5];  // [H,SI]
  const float* Wsd  = (const float*)d_in[6];  // [SI,H]
  float* out = (float*)d_out;                 // [T,H]

  // ---- workspace carve-up (~390 MB) ------------------------------------
  char* w = (char*)d_ws;
  size_t off = 0;
  auto take = [&](size_t bytes) { char* p = w + off; off = align256(off + bytes); return p; };
  __bf16* h_bf   = (__bf16*)take((size_t)TT * HH * 2);
  __bf16* Wu_bf  = (__bf16*)take((size_t)EE * HH * II * 2);
  __bf16* Wd_bf  = (__bf16*)take((size_t)EE * II * HH * 2);
  __bf16* Wsu_bf = (__bf16*)take((size_t)HH * SSI * 2);
  __bf16* Wsd_bf = (__bf16*)take((size_t)SSI * HH * 2);
  __bf16* s_act  = (__bf16*)take((size_t)TT * SSI * 2);
  __bf16* act_rt = (__bf16*)take((size_t)TT * KSEL * II * 2);
  int*    counts = (int*)take(EE * 4);
  int*    offs   = (int*)take((EE + 1) * 4);
  int*    tokens = (int*)take((size_t)EE * TT * 4);
  float*  wts    = (float*)take((size_t)EE * TT * 4);

  auto cvt = [&](const float* src, __bf16* dst, size_t n) {
    int n8 = (int)(n / 8);
    f2bf_kernel<<<(n8 + 255) / 256, 256, 0, stream>>>(src, dst, n8);
  };
  cvt(x,   h_bf,   (size_t)TT * HH);
  cvt(Wu,  Wu_bf,  (size_t)EE * HH * II);
  cvt(Wd,  Wd_bf,  (size_t)EE * II * HH);
  cvt(Wsu, Wsu_bf, (size_t)HH * SSI);
  cvt(Wsd, Wsd_bf, (size_t)SSI * HH);

  zero_counts_kernel<<<1, EE, 0, stream>>>(counts);
  router_kernel<<<TT, 32, 0, stream>>>(x, gw, eb, counts, tokens, wts);
  prefix_kernel<<<1, 1, 0, stream>>>(counts, offs);

  // shared expert: s_act = relu2(h @ Wsu)    [T,SI]
  gemm_bf16_kernel<0><<<dim3(TT / 64, SSI / 128, 1), 256, 0, stream>>>(
      h_bf, HH, Wsu_bf, SSI, TT, SSI, HH,
      nullptr, s_act, SSI, nullptr, nullptr, nullptr, nullptr);
  // out = s_act @ Wsd                        [T,H]  (plain store: init d_out)
  gemm_bf16_kernel<1><<<dim3(TT / 64, HH / 128, 1), 256, 0, stream>>>(
      s_act, SSI, Wsd_bf, HH, TT, HH, SSI,
      out, nullptr, HH, nullptr, nullptr, nullptr, nullptr);
  // routed up: act_rt[row] = relu2(h[token[row]] @ Wu[e])   [sumM, I]
  gemm_bf16_kernel<2><<<dim3(TT / 64, II / 128, EE), 256, 0, stream>>>(
      h_bf, HH, Wu_bf, II, 0, II, HH,
      nullptr, act_rt, II, counts, offs, tokens, wts);
  // routed down: out[token] += w * (act_rt[row] @ Wd[e])
  gemm_bf16_kernel<3><<<dim3(TT / 64, HH / 128, EE), 256, 0, stream>>>(
      act_rt, II, Wd_bf, HH, 0, HH, II,
      out, nullptr, HH, counts, offs, tokens, wts);
}
